// TKL_sigir20_34084860461107
// MI455X (gfx1250) — compile-verified, matrix operand order checked by reading, not run on hardware
//
#include <hip/hip_runtime.h>
#include <hip/hip_bf16.h>
#include <math.h>

// ---------------- problem constants ----------------
#define E_DIM   512
#define H_HEADS 8
#define HEAD    64
#define L_LAYERS 2
#define FF_DIM  2048
#define CHUNK   40
#define OVERLAP 5
#define EXT     50
#define SLIDE   30
#define TOPK    3
#define B_BATCH 8
#define Q_LEN   30
#define D_LEN   2000
#define NK      11
#define C_CHUNKS 50                       // chunks per batch element
#define NSEQ   (B_BATCH * C_CHUNKS + B_BATCH)   // 408 sequences (400 chunks + 8 queries)
#define SP     64                         // padded sequence length (50 and 30 -> 64)
#define MROWS  (NSEQ * SP)                // 26112 token rows
#define QKVW   (3 * E_DIM)                // 1536
#define WN     986                        // number of sliding windows

typedef __attribute__((ext_vector_type(16))) _Float16 v16h;
typedef __attribute__((ext_vector_type(8)))  _Float16 v8h;
typedef __attribute__((ext_vector_type(8)))  float    v8f;

// ---------------- WMMA fragment helpers (CDNA5 16x16x32 f16 layouts) ----------------
// A/B fragment: lane%16 = M (or N) index, lane/16 selects K half-block.
// halves j=0..7 -> K = j + hi*8 ; j=8..15 -> K = 16 + (j-8) + hi*8.
__device__ inline v16h ld_frag_rowmajor(const _Float16* __restrict__ base, int ld) {
  int lane = threadIdx.x & 31;
  int row  = lane & 15, hi = lane >> 4;
  const _Float16* p = base + (size_t)row * ld + hi * 8;
  v8h lo  = *(const v8h*)p;
  v8h hi8 = *(const v8h*)(p + 16);
  v16h r;
#pragma unroll
  for (int j = 0; j < 8; ++j) { r[j] = lo[j]; r[j + 8] = hi8[j]; }
  return r;
}

// B-matrix stored K-major: element (k, n) at base[k*ld + n]
__device__ inline v16h ld_frag_kmajor(const _Float16* __restrict__ base, int ld) {
  int lane = threadIdx.x & 31;
  int n = lane & 15, hi = lane >> 4;
  v16h r;
#pragma unroll
  for (int j = 0; j < 16; ++j) {
    int kk = ((j < 8) ? j : (j + 8)) + hi * 8;
    r[j] = base[(size_t)kk * ld + n];
  }
  return r;
}

__device__ inline v8f wmma_f16(v16h a, v16h b, v8f c) {
  return __builtin_amdgcn_wmma_f32_16x16x32_f16(false, a, false, b, (short)0, c,
                                                false, false);
}

__device__ inline float wave_sum(float v) {
#pragma unroll
  for (int off = 16; off; off >>= 1) v += __shfl_xor(v, off, 32);
  return v;
}

// ---------------- f32 -> f16 convert ----------------
__global__ void cvt_f16_kernel(const float* __restrict__ src,
                               _Float16* __restrict__ dst, size_t n) {
  size_t i = (size_t)blockIdx.x * blockDim.x + threadIdx.x;
  if (i < n) dst[i] = (_Float16)src[i];
}

// ---------------- build initial x (seq + pos), mask ----------------
__global__ __launch_bounds__(128) void build_x0_kernel(
    const float* __restrict__ qe, const float* __restrict__ de,
    const float* __restrict__ qm, const float* __restrict__ dm,
    const float* __restrict__ pos_q, const float* __restrict__ pos_d,
    _Float16* __restrict__ xf16, float* __restrict__ xres,
    float* __restrict__ mask) {
  int s = blockIdx.x >> 6;       // sequence
  int r = blockIdx.x & 63;       // padded row
  float mval = 0.f;
  const float* seqrow = nullptr;
  const float* posrow = nullptr;
  bool active = false;
  if (s < B_BATCH * C_CHUNKS) {
    int b = s / C_CHUNKS, c = s % C_CHUNKS;
    if (r < EXT) {
      int orig = c * CHUNK + r - OVERLAP;  // index into original document
      bool valid = (orig >= 0) && (orig < D_LEN);
      mval = valid ? dm[(size_t)b * D_LEN + orig] : 0.f;
      seqrow = valid ? (de + ((size_t)b * D_LEN + orig) * E_DIM) : nullptr;
      posrow = pos_d + (size_t)r * E_DIM;
      active = true;
    }
  } else {
    int qi = s - B_BATCH * C_CHUNKS;
    if (r < Q_LEN) {
      mval = qm[(size_t)qi * Q_LEN + r];
      seqrow = qe + ((size_t)qi * Q_LEN + r) * E_DIM;
      posrow = pos_q + (size_t)r * E_DIM;
      active = true;
    }
  }
  size_t rowoff = ((size_t)s * SP + r) * E_DIM;
#pragma unroll
  for (int j = 0; j < 4; ++j) {
    int e = threadIdx.x + j * 128;
    float v = 0.f;
    if (active) v = (seqrow ? seqrow[e] : 0.f) + posrow[e];
    xres[rowoff + e] = v;
    xf16[rowoff + e] = (_Float16)v;
  }
  if (threadIdx.x == 0) mask[(size_t)s * SP + r] = mval;
}

// ---------------- generic WMMA GEMM: Y = X[M,K] * W[N,K]^T + bias (+res)(relu) ----
// Epilogue shape is compile-time so the store path is branch-free straight-line code.
template <bool HAS_RES, bool RELU, bool OUT_F16>
__global__ __launch_bounds__(256) void gemm_wmma_kernel(
    const _Float16* __restrict__ X, const _Float16* __restrict__ W,
    const float* __restrict__ bias, const float* __restrict__ res,
    void* __restrict__ outp, int N, int K) {
  const int wave  = threadIdx.x >> 5;
  const int waveM = wave >> 1;                     // 0..3
  const int waveN = wave & 1;                      // 0..1
  const int m0 = blockIdx.x * 128 + waveM * 32;
  const int n0 = blockIdx.y * 128 + waveN * 64;
  v8f acc[2][4];
#pragma unroll
  for (int i = 0; i < 2; ++i)
#pragma unroll
    for (int j = 0; j < 4; ++j)
#pragma unroll
      for (int r = 0; r < 8; ++r) acc[i][j][r] = 0.f;

  for (int k0 = 0; k0 < K; k0 += 32) {
    if (k0 + 32 < K) {  // global_prefetch_b8 of next K-block through L2
      __builtin_prefetch(X + (size_t)m0 * K + k0 + 32, 0, 1);
      __builtin_prefetch(W + (size_t)n0 * K + k0 + 32, 0, 1);
    }
    v16h a[2];
    a[0] = ld_frag_rowmajor(X + (size_t)m0 * K + k0, K);
    a[1] = ld_frag_rowmajor(X + (size_t)(m0 + 16) * K + k0, K);
    v16h b[4];
#pragma unroll
    for (int j = 0; j < 4; ++j)
      b[j] = ld_frag_rowmajor(W + (size_t)(n0 + 16 * j) * K + k0, K);
#pragma unroll
    for (int i = 0; i < 2; ++i)
#pragma unroll
      for (int j = 0; j < 4; ++j) acc[i][j] = wmma_f16(a[i], b[j], acc[i][j]);
  }

  const int lane = threadIdx.x & 31;
  const int nb = lane & 15, hi = lane >> 4;
  _Float16* out16 = (_Float16*)outp;
  float*    out32 = (float*)outp;
#pragma unroll
  for (int i = 0; i < 2; ++i)
#pragma unroll
    for (int j = 0; j < 4; ++j) {
      int nn = n0 + 16 * j + nb;
      float bv = bias[nn];
#pragma unroll
      for (int r = 0; r < 8; ++r) {
        int mm = m0 + 16 * i + hi * 8 + r;
        float v = acc[i][j][r] + bv;
        if (HAS_RES) v += res[(size_t)mm * N + nn];
        if (RELU) v = v > 0.f ? v : 0.f;
        if (OUT_F16) out16[(size_t)mm * N + nn] = (_Float16)v;
        else         out32[(size_t)mm * N + nn] = v;
      }
    }
}

// ---------------- attention: one block per (sequence, head) ----------------
__global__ __launch_bounds__(256) void attn_kernel(
    const _Float16* __restrict__ qkv, const float* __restrict__ mask,
    _Float16* __restrict__ o16) {
  const int s = blockIdx.x;
  const int h = blockIdx.y;
  const _Float16* base = qkv + (size_t)s * SP * QKVW;
  const _Float16* qp = base + h * HEAD;
  const _Float16* kp = base + E_DIM + h * HEAD;
  const _Float16* vp = base + 2 * E_DIM + h * HEAD;

  __shared__ float    sc[SP][SP];
  __shared__ _Float16 a16[SP][SP + 8];

  const int wave = threadIdx.x >> 5;
  const int lane = threadIdx.x & 31;
  const int waveM = wave >> 2;  // 0..1  (32 rows each)
  const int waveN = wave & 3;   // 0..3  (16 cols each)
  const int nb = lane & 15, hi = lane >> 4;

  // scores = q @ k^T / 8 with key mask
  v8f acc[2];
#pragma unroll
  for (int i = 0; i < 2; ++i)
#pragma unroll
    for (int r = 0; r < 8; ++r) acc[i][r] = 0.f;
  for (int k0 = 0; k0 < HEAD; k0 += 32) {
    v16h b = ld_frag_rowmajor(kp + (size_t)(waveN * 16) * QKVW + k0, QKVW);
#pragma unroll
    for (int i = 0; i < 2; ++i) {
      v16h a = ld_frag_rowmajor(qp + (size_t)(waveM * 32 + i * 16) * QKVW + k0, QKVW);
      acc[i] = wmma_f16(a, b, acc[i]);
    }
  }
  const float* msk = mask + (size_t)s * SP;
#pragma unroll
  for (int i = 0; i < 2; ++i)
#pragma unroll
    for (int r = 0; r < 8; ++r) {
      int mm = waveM * 32 + i * 16 + hi * 8 + r;
      int nn = waveN * 16 + nb;
      float v = acc[i][r] * 0.125f;  // 1/sqrt(64)
      if (!(msk[nn] > 0.f)) v = -1.0e9f;
      sc[mm][nn] = v;
    }
  __syncthreads();

  // softmax: one thread per row
  if (threadIdx.x < SP) {
    int rr = threadIdx.x;
    float mx = -1e30f;
    for (int j = 0; j < SP; ++j) mx = fmaxf(mx, sc[rr][j]);
    float sum = 0.f;
    for (int j = 0; j < SP; ++j) { float e = expf(sc[rr][j] - mx); sc[rr][j] = e; sum += e; }
    float inv = 1.f / sum;
    for (int j = 0; j < SP; ++j) a16[rr][j] = (_Float16)(sc[rr][j] * inv);
  }
  __syncthreads();

  // o = a @ v
  v8f oc[2];
#pragma unroll
  for (int i = 0; i < 2; ++i)
#pragma unroll
    for (int r = 0; r < 8; ++r) oc[i][r] = 0.f;
  for (int k0 = 0; k0 < SP; k0 += 32) {
    v16h b = ld_frag_kmajor(vp + (size_t)k0 * QKVW + waveN * 16, QKVW);
#pragma unroll
    for (int i = 0; i < 2; ++i) {
      v16h a = ld_frag_rowmajor(&a16[waveM * 32 + i * 16][k0], SP + 8);
      oc[i] = wmma_f16(a, b, oc[i]);
    }
  }
#pragma unroll
  for (int i = 0; i < 2; ++i)
#pragma unroll
    for (int r = 0; r < 8; ++r) {
      int mm = waveM * 32 + i * 16 + hi * 8 + r;
      int nn = waveN * 16 + nb;
      o16[((size_t)s * SP + mm) * E_DIM + h * HEAD + nn] = (_Float16)oc[i][r];
    }
}

// ---------------- layer norm (row-wise, one wave per row) ----------------
__global__ __launch_bounds__(256) void ln_kernel(
    const float* __restrict__ y, const float* __restrict__ g,
    const float* __restrict__ beta, _Float16* __restrict__ xf16,
    float* __restrict__ xres) {
  int row = blockIdx.x * 8 + (threadIdx.x >> 5);
  int lane = threadIdx.x & 31;
  const float* yr = y + (size_t)row * E_DIM;
  float s = 0.f, s2 = 0.f;
  for (int j = lane; j < E_DIM; j += 32) { float t = yr[j]; s += t; s2 += t * t; }
  s = wave_sum(s); s2 = wave_sum(s2);
  float mean = s * (1.f / E_DIM);
  float var  = s2 * (1.f / E_DIM) - mean * mean;
  float rinv = rsqrtf(var + 1e-5f);
  for (int j = lane; j < E_DIM; j += 32) {
    float t = (yr[j] - mean) * rinv * g[j] + beta[j];
    xres[(size_t)row * E_DIM + j] = t;
    xf16[(size_t)row * E_DIM + j] = (_Float16)t;
  }
}

// ---------------- mix with original seq, mask, L2-normalize ----------------
__global__ __launch_bounds__(256) void mix_norm_kernel(
    const float* __restrict__ xres, const float* __restrict__ de,
    const float* __restrict__ qe, const float* __restrict__ mask,
    const float* __restrict__ qm, const float* __restrict__ mixer,
    _Float16* __restrict__ dn16, _Float16* __restrict__ qn16,
    float* __restrict__ m01) {
  const float mixv = mixer[0];
  const int blk = blockIdx.x;
  const int wave = threadIdx.x >> 5, lane = threadIdx.x & 31;
  __shared__ float packedsum;

  if (blk < B_BATCH * C_CHUNKS) {
    int b = blk / C_CHUNKS, c = blk % C_CHUNKS, s = blk;
    if (threadIdx.x == 0) {
      float t = 0.f;
      for (int r = OVERLAP; r < EXT - OVERLAP; ++r) t += mask[(size_t)s * SP + r];
      packedsum = t;
    }
    __syncthreads();
    float pk = (packedsum != 0.f) ? 1.f : 0.f;
    for (int r48 = wave; r48 < 48; r48 += 8) {
      _Float16* drow = dn16 + ((size_t)blk * 48 + r48) * E_DIM;
      if (r48 < CHUNK) {
        int r = r48 + OVERLAP;
        float mrow = mask[(size_t)s * SP + r];
        int orig = c * CHUNK + r48;  // always in [0, D)
        float vals[16];
        float sum2 = 0.f;
#pragma unroll
        for (int jj = 0; jj < 16; ++jj) {
          int e = lane + jj * 32;
          float sv = de[((size_t)b * D_LEN + orig) * E_DIM + e];
          float xv = xres[((size_t)s * SP + r) * E_DIM + e];
          float v = (mixv * sv + (1.f - mixv) * xv) * mrow;
          vals[jj] = v; sum2 += v * v;
        }
        sum2 = wave_sum(sum2);
        float inv = 1.f / (sqrtf(sum2) + 1e-13f);
#pragma unroll
        for (int jj = 0; jj < 16; ++jj) drow[lane + jj * 32] = (_Float16)(vals[jj] * inv);
        if (lane == 0) m01[(size_t)b * (C_CHUNKS * CHUNK) + c * CHUNK + r48] = mrow * pk;
      } else {
#pragma unroll
        for (int jj = 0; jj < 16; ++jj) drow[lane + jj * 32] = (_Float16)0.f;
      }
    }
  } else {
    int qi = blk - B_BATCH * C_CHUNKS, s = blk;
    for (int r32 = wave; r32 < 32; r32 += 8) {
      _Float16* qrow = qn16 + ((size_t)qi * 32 + r32) * E_DIM;
      if (r32 < Q_LEN) {
        float mrow = qm[(size_t)qi * Q_LEN + r32];
        float vals[16];
        float sum2 = 0.f;
#pragma unroll
        for (int jj = 0; jj < 16; ++jj) {
          int e = lane + jj * 32;
          float sv = qe[((size_t)qi * Q_LEN + r32) * E_DIM + e];
          float xv = xres[((size_t)s * SP + r32) * E_DIM + e];
          float v = (mixv * sv + (1.f - mixv) * xv) * mrow;
          vals[jj] = v; sum2 += v * v;
        }
        sum2 = wave_sum(sum2);
        float inv = 1.f / (sqrtf(sum2) + 1e-13f);
#pragma unroll
        for (int jj = 0; jj < 16; ++jj) qrow[lane + jj * 32] = (_Float16)(vals[jj] * inv);
      } else {
#pragma unroll
        for (int jj = 0; jj < 16; ++jj) qrow[lane + jj * 32] = (_Float16)0.f;
      }
    }
  }
}

// ---------------- cosine: per (b,c) 32x48x512 WMMA GEMM ----------------
__global__ __launch_bounds__(256) void cos_kernel(
    const _Float16* __restrict__ qn16, const _Float16* __restrict__ dn16,
    float* __restrict__ cosb) {
  const int bc = blockIdx.x;
  const int b = bc / C_CHUNKS;
  const int wave = threadIdx.x >> 5;
  if (wave >= 6) return;                 // wave-uniform; 2x3 fragment grid
  const int waveM = wave / 3, waveN = wave % 3;
  const _Float16* A  = qn16 + ((size_t)b * 32 + waveM * 16) * E_DIM;
  const _Float16* Bp = dn16 + ((size_t)bc * 48 + waveN * 16) * E_DIM;
  v8f acc;
#pragma unroll
  for (int r = 0; r < 8; ++r) acc[r] = 0.f;
  for (int k0 = 0; k0 < E_DIM; k0 += 32)
    acc = wmma_f16(ld_frag_rowmajor(A + k0, E_DIM),
                   ld_frag_rowmajor(Bp + k0, E_DIM), acc);
  const int lane = threadIdx.x & 31;
  const int nb = lane & 15, hi = lane >> 4;
#pragma unroll
  for (int r = 0; r < 8; ++r) {
    int q = waveM * 16 + hi * 8 + r;
    int d = waveN * 16 + nb;
    if (q < Q_LEN && d < CHUNK)
      cosb[((size_t)bc * Q_LEN + q) * CHUNK + d] = acc[r];
  }
}

// ---------------- RBF kernel pooling + window score ----------------
__global__ __launch_bounds__(128) void window_score_kernel(
    const float* __restrict__ cosb, const float* __restrict__ m01,
    const float* __restrict__ qm, const float* __restrict__ mu,
    const float* __restrict__ sigma, const float* __restrict__ dense_w,
    float* __restrict__ score) {
  const int w = blockIdx.x;
  const int b = blockIdx.y;
  __shared__ float mpos[SLIDE];
  __shared__ int   cpos[SLIDE], dpos[SLIDE];
  __shared__ float red[128];
  if (threadIdx.x < SLIDE) {
    int p = 2 * w + threadIdx.x;
    mpos[threadIdx.x] = m01[(size_t)b * (C_CHUNKS * CHUNK) + p];
    cpos[threadIdx.x] = p / CHUNK;
    dpos[threadIdx.x] = p % CHUNK;
  }
  __syncthreads();
  float anym = 0.f;
  for (int i = 0; i < SLIDE; ++i) anym += mpos[i];
  float gate = (anym > 0.f) ? 1.f : 0.f;

  float partial = 0.f;
  for (int item = threadIdx.x; item < Q_LEN * NK; item += 128) {
    int q = item / NK, kk = item % NK;
    float muv = mu[kk], sg = sigma[kk];
    float inv2s = 1.f / (2.f * sg * sg);
    float pkq = 0.f;
    for (int i = 0; i < SLIDE; ++i) {
      float m = mpos[i];
      if (m != 0.f) {
        float cv = cosb[((size_t)(b * C_CHUNKS + cpos[i]) * Q_LEN + q) * CHUNK + dpos[i]];
        float d = cv - muv;
        pkq += m * expf(-d * d * inv2s);
      }
    }
    float sat = logf(fmaxf(pkq, 1e-10f)) * qm[(size_t)b * Q_LEN + q] * gate;
    partial += sat * dense_w[kk];
  }
  red[threadIdx.x] = partial;
  __syncthreads();
  for (int st = 64; st; st >>= 1) {
    if (threadIdx.x < st) red[threadIdx.x] += red[threadIdx.x + st];
    __syncthreads();
  }
  if (threadIdx.x == 0) {
    float sc = red[0];
    score[(size_t)b * WN + w] = (sc == 0.f) ? -9000.f : sc;
  }
}

// ---------------- top-3 with suppression + neighbor scoring ----------------
__global__ __launch_bounds__(256) void topk_final_kernel(
    const float* __restrict__ score, const float* __restrict__ chunk_scoring,
    float* __restrict__ out) {
  const int b = blockIdx.x;
  __shared__ float work[WN];
  __shared__ float rv[256];
  __shared__ int   ri[256];
  __shared__ int   tops[TOPK];
  for (int i = threadIdx.x; i < WN; i += 256) work[i] = score[(size_t)b * WN + i];
  __syncthreads();
  for (int t = 0; t < TOPK; ++t) {
    float bv = -1e30f; int bi = 0;
    for (int i = threadIdx.x; i < WN; i += 256) {
      float v = work[i];
      if (v > bv) { bv = v; bi = i; }
    }
    rv[threadIdx.x] = bv; ri[threadIdx.x] = bi;
    __syncthreads();
    for (int st = 128; st; st >>= 1) {
      if (threadIdx.x < st) {
        float ov = rv[threadIdx.x + st]; int oi = ri[threadIdx.x + st];
        if (ov > rv[threadIdx.x] || (ov == rv[threadIdx.x] && oi < ri[threadIdx.x])) {
          rv[threadIdx.x] = ov; ri[threadIdx.x] = oi;
        }
      }
      __syncthreads();
    }
    if (threadIdx.x == 0) tops[t] = ri[0];
    __syncthreads();
    int best = tops[t];
    for (int i = threadIdx.x; i < WN; i += 256) {
      int dd = i > best ? i - best : best - i;
      if (dd < SLIDE / 2) work[i] = -10001.f - (float)t;
    }
    __syncthreads();
  }
  if (threadIdx.x == 0) {
    const int offs[5] = {0, -1, 1, -2, 2};
    float acc = 0.f;
    for (int j = 0; j < 5 * TOPK; ++j) {
      int id = tops[j % TOPK] + offs[j / TOPK];
      id = id < 0 ? 0 : (id > WN - 1 ? WN - 1 : id);
      float v = score[(size_t)b * WN + id];
      if (v <= -9000.f) v = 0.f;
      acc += v * chunk_scoring[j];
    }
    out[b] = acc;
  }
}

// ---------------- host launcher ----------------
extern "C" void kernel_launch(void* const* d_in, const int* in_sizes, int n_in,
                              void* d_out, int out_size, void* d_ws, size_t ws_size,
                              hipStream_t stream) {
  (void)in_sizes; (void)n_in; (void)out_size; (void)ws_size;
  const float* qe    = (const float*)d_in[0];
  const float* de    = (const float*)d_in[1];
  const float* qm    = (const float*)d_in[2];
  const float* dm    = (const float*)d_in[3];
  const float* pos_q = (const float*)d_in[4];
  const float* pos_d = (const float*)d_in[5];
  const float* Wqkv  = (const float*)d_in[6];
  const float* bqkv  = (const float*)d_in[7];
  const float* Wo    = (const float*)d_in[8];
  const float* bo    = (const float*)d_in[9];
  const float* ln1g  = (const float*)d_in[10];
  const float* ln1b  = (const float*)d_in[11];
  const float* Wff1  = (const float*)d_in[12];
  const float* bff1  = (const float*)d_in[13];
  const float* Wff2  = (const float*)d_in[14];
  const float* bff2  = (const float*)d_in[15];
  const float* ln2g  = (const float*)d_in[16];
  const float* ln2b  = (const float*)d_in[17];
  const float* mixer = (const float*)d_in[18];
  const float* dense_w = (const float*)d_in[19];
  const float* chunk_scoring = (const float*)d_in[20];
  const float* mu    = (const float*)d_in[21];
  const float* sigma = (const float*)d_in[22];
  float* out = (float*)d_out;

  // workspace carve-up
  char* ws = (char*)d_ws;
  size_t off = 0;
  auto carve = [&](size_t bytes) -> char* {
    char* p = ws + off;
    off += (bytes + 255) & ~(size_t)255;
    return p;
  };
  _Float16* w16_qkv = (_Float16*)carve((size_t)L_LAYERS * QKVW * E_DIM * 2);
  _Float16* w16_wo  = (_Float16*)carve((size_t)L_LAYERS * E_DIM * E_DIM * 2);
  _Float16* w16_ff1 = (_Float16*)carve((size_t)L_LAYERS * FF_DIM * E_DIM * 2);
  _Float16* w16_ff2 = (_Float16*)carve((size_t)L_LAYERS * E_DIM * FF_DIM * 2);
  float*    maskbuf = (float*)carve((size_t)NSEQ * SP * 4);
  _Float16* xf16    = (_Float16*)carve((size_t)MROWS * E_DIM * 2);
  float*    xres    = (float*)carve((size_t)MROWS * E_DIM * 4);
  float*    y32     = (float*)carve((size_t)MROWS * E_DIM * 4);
  char*     big     = carve((size_t)MROWS * QKVW * 2 + (size_t)MROWS * E_DIM * 2);
  _Float16* qkv16   = (_Float16*)big;
  _Float16* o16     = (_Float16*)(big + (size_t)MROWS * QKVW * 2);
  _Float16* h16     = (_Float16*)big;  // aliases qkv16+o16 (dead by FF time)
  // post-transformer buffers alias the big region too
  _Float16* dn16 = (_Float16*)big;
  _Float16* qn16 = (_Float16*)(big + (size_t)B_BATCH * C_CHUNKS * 48 * E_DIM * 2);
  float*    m01  = (float*)((char*)qn16 + (size_t)B_BATCH * 32 * E_DIM * 2);
  float*    cosb = (float*)((char*)m01 + (size_t)B_BATCH * C_CHUNKS * CHUNK * 4 + 256);
  float*    score = (float*)((char*)cosb + (size_t)B_BATCH * C_CHUNKS * Q_LEN * CHUNK * 4 + 256);

  // 1) weights -> f16
  auto cvt = [&](const float* s, _Float16* d, size_t n) {
    int blocks = (int)((n + 255) / 256);
    cvt_f16_kernel<<<blocks, 256, 0, stream>>>(s, d, n);
  };
  cvt(Wqkv, w16_qkv, (size_t)L_LAYERS * QKVW * E_DIM);
  cvt(Wo,   w16_wo,  (size_t)L_LAYERS * E_DIM * E_DIM);
  cvt(Wff1, w16_ff1, (size_t)L_LAYERS * FF_DIM * E_DIM);
  cvt(Wff2, w16_ff2, (size_t)L_LAYERS * E_DIM * FF_DIM);

  // 2) build x0 + masks
  build_x0_kernel<<<NSEQ * SP, 128, 0, stream>>>(qe, de, qm, dm, pos_q, pos_d,
                                                 xf16, xres, maskbuf);

  // 3) transformer layers
  const int MT = MROWS / 128;  // 204
  for (int l = 0; l < L_LAYERS; ++l) {
    gemm_wmma_kernel<false, false, true><<<dim3(MT, QKVW / 128), 256, 0, stream>>>(
        xf16, w16_qkv + (size_t)l * QKVW * E_DIM, bqkv + (size_t)l * QKVW,
        nullptr, qkv16, QKVW, E_DIM);
    attn_kernel<<<dim3(NSEQ, H_HEADS), 256, 0, stream>>>(qkv16, maskbuf, o16);
    gemm_wmma_kernel<true, false, false><<<dim3(MT, E_DIM / 128), 256, 0, stream>>>(
        o16, w16_wo + (size_t)l * E_DIM * E_DIM, bo + (size_t)l * E_DIM,
        xres, y32, E_DIM, E_DIM);
    ln_kernel<<<MROWS / 8, 256, 0, stream>>>(y32, ln1g + (size_t)l * E_DIM,
                                             ln1b + (size_t)l * E_DIM, xf16, xres);
    gemm_wmma_kernel<false, true, true><<<dim3(MT, FF_DIM / 128), 256, 0, stream>>>(
        xf16, w16_ff1 + (size_t)l * FF_DIM * E_DIM, bff1 + (size_t)l * FF_DIM,
        nullptr, h16, FF_DIM, E_DIM);
    gemm_wmma_kernel<true, false, false><<<dim3(MT, E_DIM / 128), 256, 0, stream>>>(
        h16, w16_ff2 + (size_t)l * E_DIM * FF_DIM, bff2 + (size_t)l * E_DIM,
        xres, y32, E_DIM, FF_DIM);
    ln_kernel<<<MROWS / 8, 256, 0, stream>>>(y32, ln2g + (size_t)l * E_DIM,
                                             ln2b + (size_t)l * E_DIM, xf16, xres);
  }

  // 4) mix + normalize (writes dn16/qn16/m01 into region where qkv/h are dead)
  mix_norm_kernel<<<NSEQ, 256, 0, stream>>>(xres, de, qe, maskbuf, qm, mixer,
                                            dn16, qn16, m01);

  // 5) cosine similarities
  cos_kernel<<<B_BATCH * C_CHUNKS, 256, 0, stream>>>(qn16, dn16, cosb);

  // 6) window scores
  window_score_kernel<<<dim3(WN, B_BATCH), 128, 0, stream>>>(
      cosb, m01, qm, mu, sigma, dense_w, score);

  // 7) top-k + final
  topk_final_kernel<<<B_BATCH, 256, 0, stream>>>(score, chunk_scoring, out);
}
